// CharLSTM_75814762709790
// MI455X (gfx1250) — compile-verified, matrix operand order checked by reading, not run on hardware
//
#include <hip/hip_runtime.h>

// ---------------- problem constants ----------------
#define N_CHARS 256
#define EMB     512
#define HID     2048
#define TLEN    8192
#define KZ      (EMB + HID)        // 2560 = z-vector length
#define G4      (4 * HID)          // 8192 gate outputs
#define TILES_G (G4 / 16)          // 512 16-row tiles
#define KSTEPS_G (KZ / 32)         // 80 WMMA k-steps (K=32 each)
#define TILES_Y (N_CHARS / 16)     // 16
#define KSTEPS_Y (HID / 32)        // 64
#define NB      64                 // persistent grid (blocks)
#define TPB     256                // 8 waves/block (wave32)
#define WPB     (TPB / 32)

typedef __attribute__((ext_vector_type(16))) __bf16 v16bf;
typedef __attribute__((ext_vector_type(8)))  float  v8f;

__device__ __forceinline__ __bf16 f2bf(float f) {
  union { float f; unsigned u; } c; c.f = f;
  unsigned u = c.u + 0x7FFFu + ((c.u >> 16) & 1u);   // round-to-nearest-even
  unsigned short s = (unsigned short)(u >> 16);
  __bf16 r; __builtin_memcpy(&r, &s, 2); return r;
}

__device__ __forceinline__ float fast_sigmoid(float x) {
  return __builtin_amdgcn_rcpf(1.0f + __expf(-x));
}
__device__ __forceinline__ float fast_tanh(float x) {
  return 1.0f - 2.0f * __builtin_amdgcn_rcpf(__expf(2.0f * x) + 1.0f);
}

// ---------------- packing: W -> WMMA A-fragment bf16 layout ----------------
// Fragment = 16x32 bf16 tile, 512 elems. Per ISA 7.12.2 (16-bit A, wave32):
//   lane 0-15 : M = lane,    elems 0-7 -> K 0..7,  elems 8-15 -> K 16..23
//   lane16-31 : M = lane-16, elems 0-7 -> K 8..15, elems 8-15 -> K 24..31
__global__ void CharLSTM_pack_gates(const float* __restrict__ Wf,
                                    const float* __restrict__ Wi,
                                    const float* __restrict__ Wc,
                                    const float* __restrict__ Wo,
                                    __bf16* __restrict__ out) {
  size_t idx = (size_t)blockIdx.x * TPB + threadIdx.x;     // total = 512*80*512
  int e    = (int)(idx & 15);
  int lane = (int)((idx >> 4) & 31);
  size_t frag = idx >> 9;
  int ks    = (int)(frag % KSTEPS_G);
  int gtile = (int)(frag / KSTEPS_G);
  int row = gtile * 16 + (lane & 15);                      // 0..8191 in [f;i;c;o]
  int k   = ks * 32 + (e & 7) + ((lane >> 4) << 3) + ((e >> 3) << 4);
  int sel = row >> 11;
  const float* W = (sel == 0) ? Wf : (sel == 1) ? Wi : (sel == 2) ? Wc : Wo;
  out[idx] = f2bf(W[(size_t)(row & (HID - 1)) * KZ + k]);
}

__global__ void CharLSTM_pack_wy(const float* __restrict__ Wy,
                                 __bf16* __restrict__ out) {
  size_t idx = (size_t)blockIdx.x * TPB + threadIdx.x;     // total = 16*64*512
  int e    = (int)(idx & 15);
  int lane = (int)((idx >> 4) & 31);
  size_t frag = idx >> 9;
  int ks   = (int)(frag % KSTEPS_Y);
  int tile = (int)(frag / KSTEPS_Y);
  int row = tile * 16 + (lane & 15);
  int k   = ks * 32 + (e & 7) + ((lane >> 4) << 3) + ((e >> 3) << 4);
  out[idx] = f2bf(Wy[(size_t)row * HID + k]);
}

__global__ void CharLSTM_pack_bias(const float* __restrict__ bf,
                                   const float* __restrict__ bi,
                                   const float* __restrict__ bc,
                                   const float* __restrict__ bo,
                                   float* __restrict__ ball) {
  int i = blockIdx.x * TPB + threadIdx.x;
  if (i < G4) {
    int s = i >> 11;
    const float* b = (s == 0) ? bf : (s == 1) ? bi : (s == 2) ? bc : bo;
    ball[i] = b[i & (HID - 1)];
  }
}

__global__ void CharLSTM_embed(const int* __restrict__ seq,
                               const float* __restrict__ emb,
                               __bf16* __restrict__ xbf) {
  size_t idx = (size_t)blockIdx.x * TPB + threadIdx.x;     // total = TLEN*EMB
  int t = (int)(idx >> 9);
  int j = (int)(idx & (EMB - 1));
  xbf[idx] = f2bf(emb[(size_t)seq[t] * EMB + j]);
}

__global__ void CharLSTM_init(__bf16* __restrict__ hbf, float* __restrict__ hf,
                              float* __restrict__ cf, unsigned* __restrict__ cnt,
                              __bf16* __restrict__ zpad) {
  int i = blockIdx.x * TPB + threadIdx.x;
  if (i < HID) { hbf[i] = f2bf(0.0f); hf[i] = 0.0f; cf[i] = 0.0f; }
  if (i < 32)  { zpad[i] = f2bf(0.0f); }
  if (i == 0)  { *cnt = 0u; }
}

// ---------------- grid-wide barrier (monotone counter) ----------------
__device__ __forceinline__ void grid_sync(volatile unsigned* cnt, unsigned expected) {
  __syncthreads();
  if (threadIdx.x == 0) {
    __threadfence();
    atomicAdd((unsigned*)cnt, 1u);
    while (*cnt < expected) { __builtin_amdgcn_s_sleep(1); }
    __threadfence();
  }
  __syncthreads();
}

// ---------------- persistent recurrent kernel ----------------
__global__ __launch_bounds__(TPB, 1) void CharLSTM_persist(
    const __bf16* __restrict__ WG,  const __bf16* __restrict__ WY,
    const __bf16* __restrict__ XBF, const float*  __restrict__ BALL,
    const float*  __restrict__ BY,  float* __restrict__ GBUF,
    __bf16* __restrict__ HBF, float* __restrict__ HF, float* __restrict__ CF,
    unsigned* __restrict__ CNT, const __bf16* __restrict__ ZPAD,
    float* __restrict__ OUT) {
  // z = [x_t ; h] in bf16, plus a 32-element zero pad that the 30 non-column-0
  // lanes read from (stride 0) so the B-fragment build has NO divergence and
  // NO per-kstep zero-fills.
  __shared__ __bf16 zsh[KZ + 32];
  const int tid  = threadIdx.x;
  const int lane = tid & 31;
  const int wave = tid >> 5;
  const int blk  = blockIdx.x;
  const int gwave = blk * WPB + wave;            // 0..511, == gate tile id
  unsigned bar = 0;

  if (tid < 32) zsh[KZ + tid] = f2bf(0.0f);      // zero pad, written once

  // Lane-constant B-source pointers: lanes with n==0 stream z/h, others a pad.
  const bool bl = ((lane & 15) == 0);
  const int  zoff    = bl ? ((lane >> 4) << 3) : KZ;   // LDS start index
  const int  zstride = bl ? 32 : 0;
  const __bf16* hbase   = bl ? (HBF + ((lane >> 4) << 3)) : ZPAD;
  const int     hstride = bl ? 32 : 0;

  for (int t = 0; t < TLEN; ++t) {
    // ---- phase 0: stage z into LDS
    for (int i = tid; i < KZ; i += TPB)
      zsh[i] = (i < EMB) ? XBF[(size_t)t * EMB + i] : HBF[i - EMB];
    __syncthreads();

    // ---- phase 1: gate GEMV tile (all 512 waves), f32 accumulate
    {
      const __bf16* wbase = WG + (((size_t)gwave * KSTEPS_G) << 9);
      v8f acc = {};
      for (int ks = 0; ks < KSTEPS_G; ++ks) {
        const __bf16* fr = wbase + ((size_t)ks << 9) + (lane << 4);
        v16bf a = *(const v16bf*)fr;             // A fragment: 32B/lane
        __builtin_prefetch(fr + 512, 0, 1);      // global_prefetch_b8 next frag
        const __bf16* zp = zsh + zoff + ks * zstride;
        v16bf b;                                 // fully written below (or pad)
#pragma unroll
        for (int e = 0; e < 8; ++e) { b[e] = zp[e]; b[e + 8] = zp[e + 16]; }
        acc = __builtin_amdgcn_wmma_f32_16x16x32_bf16(
            false, a, false, b, (short)0, acc, false, false);
      }
      if ((lane & 15) == 0) {                    // D column 0 -> lanes 0 & 16
        int mb = gwave * 16 + ((lane >> 4) << 3);
#pragma unroll
        for (int r = 0; r < 8; ++r) GBUF[mb + r] = acc[r] + BALL[mb + r];
      }
    }
    grid_sync(CNT, ++bar * NB);

    // ---- phase 2: gate nonlinearity + state update (blocks 0..7 cover 2048)
    if (blk < 8) {
      int i = blk * TPB + tid;
      float gf = GBUF[i], gi = GBUF[HID + i];
      float gc = GBUF[2 * HID + i], go = GBUF[3 * HID + i];
      float ft = fast_sigmoid(gf);
      float it = fast_sigmoid(gi);
      float ct = fast_tanh(gc);
      float ot = fast_sigmoid(go);
      float c = ft * CF[i] + it * ct;
      float h = ot * fast_tanh(c);
      CF[i] = c; HF[i] = h; HBF[i] = f2bf(h);
    }
    grid_sync(CNT, ++bar * NB);

    // ---- phase 3: y_t = W_y @ h_t + b_y (waves 0..15)
    if (gwave < TILES_Y) {
      const __bf16* wbase = WY + (((size_t)gwave * KSTEPS_Y) << 9);
      v8f acc = {};
      for (int ks = 0; ks < KSTEPS_Y; ++ks) {
        const __bf16* fr = wbase + ((size_t)ks << 9) + (lane << 4);
        v16bf a = *(const v16bf*)fr;
        const __bf16* hp = hbase + ks * hstride;
        v16bf b;
#pragma unroll
        for (int e = 0; e < 8; ++e) { b[e] = hp[e]; b[e + 8] = hp[e + 16]; }
        acc = __builtin_amdgcn_wmma_f32_16x16x32_bf16(
            false, a, false, b, (short)0, acc, false, false);
      }
      if ((lane & 15) == 0) {
        int rb = gwave * 16 + ((lane >> 4) << 3);
#pragma unroll
        for (int r = 0; r < 8; ++r)
          OUT[(size_t)t * N_CHARS + rb + r] = acc[r] + BY[rb + r];
      }
    }
  }

  // ---- epilogue: final h, c (same threads that wrote them)
  if (blk < 8) {
    int i = blk * TPB + tid;
    OUT[(size_t)TLEN * N_CHARS + i] = HF[i];
    OUT[(size_t)TLEN * N_CHARS + HID + i] = CF[i];
  }
}

// ---------------- host launcher ----------------
extern "C" void kernel_launch(void* const* d_in, const int* in_sizes, int n_in,
                              void* d_out, int out_size, void* d_ws, size_t ws_size,
                              hipStream_t stream) {
  const int*   seq = (const int*)d_in[0];
  const float* emb = (const float*)d_in[1];
  const float* Wf  = (const float*)d_in[2];
  const float* bf  = (const float*)d_in[3];
  const float* Wi  = (const float*)d_in[4];
  const float* bi  = (const float*)d_in[5];
  const float* Wo  = (const float*)d_in[6];
  const float* bo  = (const float*)d_in[7];
  const float* Wc  = (const float*)d_in[8];
  const float* bc  = (const float*)d_in[9];
  const float* Wy  = (const float*)d_in[10];
  const float* by  = (const float*)d_in[11];
  float* out = (float*)d_out;

  char* ws = (char*)d_ws;
  auto carve = [&](size_t bytes) -> char* {
    char* p = ws; ws += (bytes + 255) & ~(size_t)255; return p;
  };
  __bf16*   WG   = (__bf16*)carve((size_t)TILES_G * KSTEPS_G * 512 * 2); // 40 MB
  __bf16*   WYp  = (__bf16*)carve((size_t)TILES_Y * KSTEPS_Y * 512 * 2); //  1 MB
  __bf16*   XBF  = (__bf16*)carve((size_t)TLEN * EMB * 2);               //  8 MB
  float*    BALL = (float*)carve((size_t)G4 * 4);
  float*    GBUF = (float*)carve((size_t)G4 * 4);
  __bf16*   HBF  = (__bf16*)carve((size_t)HID * 2);
  float*    HF   = (float*)carve((size_t)HID * 4);
  float*    CF   = (float*)carve((size_t)HID * 4);
  unsigned* CNT  = (unsigned*)carve(256);
  __bf16*   ZPAD = (__bf16*)carve(256);

  CharLSTM_init<<<(HID + TPB - 1) / TPB, TPB, 0, stream>>>(HBF, HF, CF, CNT, ZPAD);
  CharLSTM_pack_bias<<<(G4 + TPB - 1) / TPB, TPB, 0, stream>>>(bf, bi, bc, bo, BALL);
  CharLSTM_embed<<<((size_t)TLEN * EMB) / TPB, TPB, 0, stream>>>(seq, emb, XBF);
  CharLSTM_pack_gates<<<((size_t)TILES_G * KSTEPS_G * 512) / TPB, TPB, 0, stream>>>(
      Wf, Wi, Wc, Wo, WG);
  CharLSTM_pack_wy<<<((size_t)TILES_Y * KSTEPS_Y * 512) / TPB, TPB, 0, stream>>>(Wy, WYp);
  CharLSTM_persist<<<NB, TPB, 0, stream>>>(WG, WYp, XBF, BALL, by, GBUF,
                                           HBF, HF, CF, CNT, ZPAD, out);
}
